// LVPruneController_36661840838747
// MI455X (gfx1250) — matrix operand order repaired and checked by reading, not compiled, not run on hardware
//
#include <hip/hip_runtime.h>
#include <cstddef>
#include <cstdint>

// ---------------------------------------------------------------------------
// Problem constants
// ---------------------------------------------------------------------------
constexpr int B_   = 128;
constexpr int L_   = 197;
constexpr int C_   = 768;
constexpr int Lt_  = 32;
constexpr int Ct_  = 512;
constexpr int H_   = 12;
constexpr int DH_  = 64;
constexpr int DPP_ = 256;
constexpr int Nv_  = 196;                 // L-1
constexpr int ROWS = B_ * Nv_;            // 25088 (multiple of 16)
constexpr float TARGET_ = 0.7f;
constexpr float G_SCALE = 1.0f / (float(DPP_) * float(Nv_) * 0.01f); // 1/501.76
constexpr float DPP_EPS = 1e-5f;

typedef __bf16 bf16;
typedef __attribute__((ext_vector_type(16))) __bf16 v16bf;
typedef __attribute__((ext_vector_type(8)))  __bf16 v8bf;
typedef __attribute__((ext_vector_type(8)))  float  v8f;

#define DEV static __device__ __forceinline__

// ---------------------------------------------------------------------------
// WMMA fragment loaders (16x16x32 bf16; this code's contiguous k order, same for A and B, not the ISA table order)
// A (16x32, MxK): lane m<16 holds row m, K 0..15; lane m+16 holds
//                 row m, K 16..31.  -> two 16B loads per lane.
// B (32x16, KxN): lane holds column n = lane&15; lanes 0-15 carry K 0..15,
//                 lanes 16-31 carry K 16..31. -> one contiguous 32B row
//                 slice of W (since B[k][n] = W[n][k]).
// ---------------------------------------------------------------------------
DEV v16bf zero_frag() {
  v16bf r;
#pragma unroll
  for (int i = 0; i < 16; ++i) r[i] = (bf16)0.0f;
  return r;
}

DEV v16bf load_a_frag(const bf16* __restrict__ A, int lda, int row, int k0,
                      int lane, bool valid) {
  if (!valid) return zero_frag();
  int kb = k0 + ((lane >> 4) << 4);
  const bf16* p = A + (size_t)row * lda + kb;
  v8bf lo = *(const v8bf*)(p);
  v8bf hi = *(const v8bf*)(p + 8);
  v16bf r;
#pragma unroll
  for (int i = 0; i < 8; ++i) { r[i] = lo[i]; r[i + 8] = hi[i]; }
  return r;
}

DEV v16bf load_b_frag(const bf16* __restrict__ W, int ldw, int col, int k0,
                      int lane, bool valid) {
  if (!valid) return zero_frag();
  int kb = k0 + ((lane >> 4) << 4);
  return *(const v16bf*)(W + (size_t)col * ldw + kb);
}

// ---------------------------------------------------------------------------
// Generic bf16 WMMA GEMM:  Out[M,N] = A[M,K] @ W[N,K]^T (+bias)(+act)
// Each wave computes a 16x64 tile (4 accumulators): the A fragment is loaded
// once per k-step and reused across 4 WMMAs, cutting big-operand (A) traffic
// 4x vs one-tile-per-wave. 8 waves / 256-thread block.
// Requires: M % 16 == 0, N % 64 == 0, K % 32 == 0.
// act: 0 = none, 1 = x*sigmoid(1.702x)
// ---------------------------------------------------------------------------
__global__ __launch_bounds__(256)
void gemm_bf16_wmma(const bf16* __restrict__ A, int lda,
                    const bf16* __restrict__ W, int ldw,
                    const float* __restrict__ bias,
                    float* __restrict__ outF, bf16* __restrict__ outBf,
                    int ldo, int M, int N, int K, int act) {
  int wave = threadIdx.x >> 5, lane = threadIdx.x & 31;
  int ngroups = N >> 6;                              // 64-wide N groups
  int tile = blockIdx.x * 8 + wave;
  if (tile >= (M >> 4) * ngroups) return;
  int m0 = (tile / ngroups) << 4;
  int n0 = (tile % ngroups) << 6;

  v8f acc[4];
#pragma unroll
  for (int j = 0; j < 4; ++j)
    acc[j] = (v8f){0.f, 0.f, 0.f, 0.f, 0.f, 0.f, 0.f, 0.f};

  int arow = m0 + (lane & 15);
  int bcol = n0 + (lane & 15);
  for (int k0 = 0; k0 < K; k0 += 32) {
    v16bf a = load_a_frag(A, lda, arow, k0, lane, true);
#pragma unroll
    for (int j = 0; j < 4; ++j) {
      v16bf b = load_b_frag(W, ldw, bcol + 16 * j, k0, lane, true);
      acc[j] = __builtin_amdgcn_wmma_f32_16x16x32_bf16(false, a, false, b,
                                                       (short)0, acc[j],
                                                       false, false);
    }
  }
  int mb = m0 + ((lane >> 4) << 3);
#pragma unroll
  for (int j = 0; j < 4; ++j) {
    int n = n0 + 16 * j + (lane & 15);
    float bv = bias ? bias[n] : 0.0f;
#pragma unroll
    for (int i = 0; i < 8; ++i) {
      float val = acc[j][i] + bv;
      if (act == 1) val = val / (1.0f + __expf(-1.702f * val));
      size_t o = (size_t)(mb + i) * ldo + n;
      if (outF) outF[o] = val;
      else      outBf[o] = (bf16)val;
    }
  }
}

// ---------------------------------------------------------------------------
// Per-batch Gram matrix: G[b] = pw[b](196x256) @ pw[b]^T * G_SCALE + (1+eps)I
// 13x13 tile grid with zero-padded fragments beyond row/col 195.
// ---------------------------------------------------------------------------
__global__ __launch_bounds__(256)
void gram_wmma(const bf16* __restrict__ pw, float* __restrict__ G) {
  int b = blockIdx.x;
  int wave = threadIdx.x >> 5, lane = threadIdx.x & 31;
  constexpr int T = 13;                        // ceil(196/16)
  int tile = blockIdx.y * 8 + wave;
  if (tile >= T * T) return;
  int m0 = (tile / T) << 4, n0 = (tile % T) << 4;
  const bf16* P = pw + (size_t)b * Nv_ * DPP_;

  v8f acc = {0.f, 0.f, 0.f, 0.f, 0.f, 0.f, 0.f, 0.f};
  int arow = m0 + (lane & 15);
  int bcol = n0 + (lane & 15);
  for (int k0 = 0; k0 < DPP_; k0 += 32) {
    v16bf a  = load_a_frag(P, DPP_, arow, k0, lane, arow < Nv_);
    v16bf bm = load_b_frag(P, DPP_, bcol, k0, lane, bcol < Nv_);
    acc = __builtin_amdgcn_wmma_f32_16x16x32_bf16(false, a, false, bm,
                                                  (short)0, acc, false, false);
  }
  int n  = n0 + (lane & 15);
  int mb = m0 + ((lane >> 4) << 3);
  if (n >= Nv_) return;
  float* Gb = G + (size_t)b * Nv_ * Nv_;
#pragma unroll
  for (int i = 0; i < 8; ++i) {
    int m = mb + i;
    if (m < Nv_) {
      float v = acc[i] * G_SCALE;
      if (m == n) v += 1.0f + DPP_EPS;
      Gb[(size_t)m * Nv_ + n] = v;
    }
  }
}

// ---------------------------------------------------------------------------
// Converters
// ---------------------------------------------------------------------------
__global__ void convert_vis(const float* __restrict__ x, bf16* __restrict__ vis) {
  size_t total = (size_t)ROWS * C_;
  for (size_t i = (size_t)blockIdx.x * 256 + threadIdx.x; i < total;
       i += (size_t)gridDim.x * 256) {
    size_t c = i % C_;
    size_t r = i / C_;
    size_t b = r / Nv_, n = r % Nv_;
    vis[i] = (bf16)x[((n + 1) * (size_t)B_ + b) * C_ + c];   // x_LND[n+1, b, c]
  }
}

__global__ void convert_f2bf(const float* __restrict__ src, bf16* __restrict__ dst,
                             size_t n) {
  for (size_t i = (size_t)blockIdx.x * 256 + threadIdx.x; i < n;
       i += (size_t)gridDim.x * 256)
    dst[i] = (bf16)src[i];
}

__global__ void zero_sums(float* s) { if (threadIdx.x < 2) s[threadIdx.x] = 0.f; }

// ---------------------------------------------------------------------------
// Attention: grid (H, B), 256 threads. K/V tile (32x64) staged in LDS.
// One lane per text token (wave32 == Lt), shuffle softmax, 2 ctx dims/lane.
// Writes fused = ctx + vis_tok directly in bf16 for the hmid GEMM.
// rel_BN is analytically 1/Lt (softmax rows sum to 1) -> rel == 1: no stats.
// ---------------------------------------------------------------------------
__global__ __launch_bounds__(256)
void attn_kernel(const float* __restrict__ q, const float* __restrict__ kbuf,
                 const float* __restrict__ vbuf, const float* __restrict__ x,
                 bf16* __restrict__ fused) {
  int h = blockIdx.x, b = blockIdx.y;
  __shared__ float ks[Lt_][DH_ + 1];
  __shared__ float vs[Lt_][DH_ + 1];
  int tid = threadIdx.x;
  for (int idx = tid; idx < Lt_ * DH_; idx += 256) {
    int l = idx >> 6, d = idx & 63;
    size_t base = ((size_t)(b * Lt_ + l)) * C_ + h * DH_ + d;
    ks[l][d] = kbuf[base];
    vs[l][d] = vbuf[base];
  }
  __syncthreads();
  int wave = tid >> 5, lane = tid & 31;
  for (int n = wave; n < Nv_; n += 8) {
    const float* qr = q + ((size_t)(b * Nv_ + n)) * C_ + h * DH_;
    float s = 0.f;
#pragma unroll
    for (int d = 0; d < DH_; ++d) s += qr[d] * ks[lane][d];
    s *= 0.125f;                                      // 1/sqrt(DH)
    float mx = s;
#pragma unroll
    for (int m = 16; m; m >>= 1) mx = fmaxf(mx, __shfl_xor(mx, m, 32));
    float e = __expf(s - mx);
    float sum = e;
#pragma unroll
    for (int m = 16; m; m >>= 1) sum += __shfl_xor(sum, m, 32);
    float a = e / sum;

    float acc0 = 0.f, acc1 = 0.f;
    int d0 = lane * 2;
#pragma unroll
    for (int l = 0; l < Lt_; ++l) {
      float al = __shfl(a, l, 32);
      acc0 += al * vs[l][d0];
      acc1 += al * vs[l][d0 + 1];
    }
    int c0 = h * DH_ + d0;
    const float* vt = x + ((size_t)(n + 1) * B_ + b) * C_ + c0;   // vis_tok
    bf16* o = fused + ((size_t)(b * Nv_ + n)) * C_ + c0;
    o[0] = (bf16)(acc0 + vt[0]);
    o[1] = (bf16)(acc1 + vt[1]);
  }
}

// ---------------------------------------------------------------------------
// logits (2-wide) + gumbel-softmax + policy + keep_prob + sum for loss_ratio
// One wave per row (8 rows per block).
// ---------------------------------------------------------------------------
__global__ __launch_bounds__(256)
void logits_kernel(const bf16* __restrict__ hmid, const float* __restrict__ W2,
                   const float* __restrict__ b2, const float* __restrict__ gum,
                   float* __restrict__ policy, float* __restrict__ keep_prob,
                   float* __restrict__ sums) {
  int wave = threadIdx.x >> 5, lane = threadIdx.x & 31;
  int row = blockIdx.x * 8 + wave;
  const bf16* hr = hmid + (size_t)row * C_;
  float s0 = 0.f, s1 = 0.f;
  for (int c = lane; c < C_; c += 32) {
    float h = (float)hr[c];
    s0 += h * W2[c];
    s1 += h * W2[C_ + c];
  }
#pragma unroll
  for (int m = 16; m; m >>= 1) {
    s0 += __shfl_xor(s0, m, 32);
    s1 += __shfl_xor(s1, m, 32);
  }
  if (lane == 0) {
    float l0 = s0 + b2[0], l1 = s1 + b2[1];
    int b = row / Nv_, n = row % Nv_;
    float u0 = fminf(fmaxf(gum[(size_t)row * 2 + 0], 1e-6f), 1.f - 1e-6f);
    float u1 = fminf(fmaxf(gum[(size_t)row * 2 + 1], 1e-6f), 1.f - 1e-6f);
    float g0 = -logf(-logf(u0));
    float g1 = -logf(-logf(u1));
    float mxl = fmaxf(l0, l1);
    float y0 = (l0 - mxl) + g0;                       // TAU == 1
    float y1 = (l1 - mxl) + g1;
    float m2 = fmaxf(y0, y1);
    float e0 = __expf(y0 - m2), e1 = __expf(y1 - m2);
    float p1 = e1 / (e0 + e1);
    policy[(size_t)b * L_ + 1 + n] = (y1 > y0) ? 1.0f : 0.0f; // argmax, tie->0
    if (n == 0) policy[(size_t)b * L_] = 1.0f;                 // CLS kept
    keep_prob[row] = p1;
    atomicAdd(&sums[0], p1);
  }
}

// ---------------------------------------------------------------------------
// Per-batch DPP weight: w = sqrt(max(kp / (mean_b(kp)+eps), 1e-6))  (rel==1)
// ---------------------------------------------------------------------------
__global__ __launch_bounds__(256)
void w_kernel(const float* __restrict__ keep_prob, float* __restrict__ wout) {
  int b = blockIdx.x, tid = threadIdx.x;
  __shared__ float red[256];
  float s = 0.f;
  for (int n = tid; n < Nv_; n += 256)
    s += fminf(fmaxf(keep_prob[b * Nv_ + n], 0.f), 1.f);
  red[tid] = s;
  __syncthreads();
  for (int st = 128; st; st >>= 1) {
    if (tid < st) red[tid] += red[tid + st];
    __syncthreads();
  }
  float mean = red[0] / (float)Nv_;
  for (int n = tid; n < Nv_; n += 256) {
    float kp = fminf(fmaxf(keep_prob[b * Nv_ + n], 0.f), 1.f);
    float w = fmaxf(kp / (mean + 1e-12f), 1e-6f);
    wout[b * Nv_ + n] = sqrtf(w);
  }
}

// ---------------------------------------------------------------------------
// Row-normalize phi and scale by w; emit bf16 pw for the Gram WMMA.
// ---------------------------------------------------------------------------
__global__ __launch_bounds__(256)
void pw_kernel(const float* __restrict__ phi, const float* __restrict__ wrow,
               bf16* __restrict__ pw) {
  int wave = threadIdx.x >> 5, lane = threadIdx.x & 31;
  int row = blockIdx.x * 8 + wave;
  const float* pr = phi + (size_t)row * DPP_;
  float ss = 0.f;
  for (int j = lane; j < DPP_; j += 32) { float v = pr[j]; ss += v * v; }
#pragma unroll
  for (int m = 16; m; m >>= 1) ss += __shfl_xor(ss, m, 32);
  float sc = wrow[row] / fmaxf(sqrtf(ss), 1e-12f);
  for (int j = lane; j < DPP_; j += 32)
    pw[(size_t)row * DPP_ + j] = (bf16)(pr[j] * sc);
}

// ---------------------------------------------------------------------------
// In-place Cholesky of 196x196 per batch (G lives in L2: 19.7MB << 192MB),
// right-looking, then logdet accumulation.
// ---------------------------------------------------------------------------
__global__ __launch_bounds__(256)
void cholesky_kernel(float* __restrict__ G, float* __restrict__ sums) {
  int b = blockIdx.x, tid = threadIdx.x;
  float* A = G + (size_t)b * Nv_ * Nv_;
  __shared__ float dsh;
  for (int k = 0; k < Nv_; ++k) {
    if (tid == 0) {
      float d = sqrtf(A[(size_t)k * Nv_ + k]);
      A[(size_t)k * Nv_ + k] = d;
      dsh = d;
    }
    __syncthreads();
    float dinv = 1.0f / dsh;
    for (int i = k + 1 + tid; i < Nv_; i += 256) A[(size_t)i * Nv_ + k] *= dinv;
    __syncthreads();
    int rem = Nv_ - 1 - k;                 // rows/cols k+1..195
    int total = rem * rem;                 // symmetric update of both halves
    for (int t = tid; t < total; t += 256) {
      int i = k + 1 + t / rem;
      int j = k + 1 + t % rem;
      A[(size_t)i * Nv_ + j] -= A[(size_t)i * Nv_ + k] * A[(size_t)j * Nv_ + k];
    }
    __syncthreads();
  }
  __shared__ float red[256];
  float s = 0.f;
  for (int k = tid; k < Nv_; k += 256) s += logf(A[(size_t)k * Nv_ + k]);
  red[tid] = s;
  __syncthreads();
  for (int st = 128; st; st >>= 1) {
    if (tid < st) red[tid] += red[tid + st];
    __syncthreads();
  }
  if (tid == 0) atomicAdd(&sums[1], 2.0f * red[0]);
}

__global__ void finalize_kernel(const float* __restrict__ sums,
                                float* __restrict__ out) {
  float d = sums[0] / (float)ROWS - TARGET_;
  out[B_ * L_]     = d * d;                  // loss_ratio
  out[B_ * L_ + 1] = -(sums[1] / (float)B_); // loss_dpp
}

// ---------------------------------------------------------------------------
// Host launcher
// ---------------------------------------------------------------------------
extern "C" void kernel_launch(void* const* d_in, const int* in_sizes, int n_in,
                              void* d_out, int out_size, void* d_ws, size_t ws_size,
                              hipStream_t stream) {
  (void)in_sizes; (void)n_in; (void)out_size; (void)ws_size;
  const float* x    = (const float*)d_in[0];   // [L,B,C]
  const float* text = (const float*)d_in[1];   // [B,Lt,Ct]
  const float* gum  = (const float*)d_in[2];   // [B,Nv,2]
  const float* Wq = (const float*)d_in[3];  const float* bq = (const float*)d_in[4];
  const float* Wk = (const float*)d_in[5];  const float* bk = (const float*)d_in[6];
  const float* Wv = (const float*)d_in[7];  const float* bv = (const float*)d_in[8];
  const float* W1 = (const float*)d_in[9];  const float* b1 = (const float*)d_in[10];
  const float* W2 = (const float*)d_in[11]; const float* b2 = (const float*)d_in[12];
  const float* Wdpp = (const float*)d_in[13];
  float* out = (float*)d_out;                  // [policy 25216 | loss_ratio | loss_dpp]

  // ---- workspace carve-up (256B aligned), with phase-based overlays ----
  char* ws = (char*)d_ws;
  size_t off = 0;
  auto alloc = [&](size_t bytes) {
    size_t o = off;
    off += (bytes + 255) & ~size_t(255);
    return o;
  };
  bf16* vis_bf  = (bf16*)(ws + alloc((size_t)ROWS * C_ * 2));
  bf16* text_bf = (bf16*)(ws + alloc((size_t)B_ * Lt_ * Ct_ * 2));
  bf16* Wq_bf   = (bf16*)(ws + alloc((size_t)C_ * C_ * 2));
  bf16* Wk_bf   = (bf16*)(ws + alloc((size_t)C_ * Ct_ * 2));
  bf16* Wv_bf   = (bf16*)(ws + alloc((size_t)C_ * Ct_ * 2));
  bf16* W1_bf   = (bf16*)(ws + alloc((size_t)C_ * C_ * 2));
  bf16* Wdpp_bf = (bf16*)(ws + alloc((size_t)DPP_ * C_ * 2));
  float* q_f    = (float*)(ws + alloc((size_t)ROWS * C_ * 4));   // 77MB
  float* k_f    = (float*)(ws + alloc((size_t)B_ * Lt_ * C_ * 4));
  float* v_f    = (float*)(ws + alloc((size_t)B_ * Lt_ * C_ * 4));
  bf16* fused_bf = (bf16*)(ws + alloc((size_t)ROWS * C_ * 2));
  bf16* hmid_bf  = (bf16*)(ws + alloc((size_t)ROWS * C_ * 2));
  float* keep_prob = (float*)(ws + alloc((size_t)ROWS * 4));
  float* wrow      = (float*)(ws + alloc((size_t)ROWS * 4));
  float* sums      = (float*)(ws + alloc(2 * 4));
  // overlays: phi+pw reuse q (dead after attention); G reuses k/v (dead too)
  float* phi  = q_f;                                          // 25.7MB
  bf16*  pw   = (bf16*)((char*)q_f +
                        (((size_t)ROWS * DPP_ * 4 + 255) & ~size_t(255)));
  float* Gm   = k_f;                                          // 19.7MB <= k+v

  auto cdiv = [](int a, int b) { return (a + b - 1) / b; };

  // 0) zero accumulators
  zero_sums<<<1, 32, 0, stream>>>(sums);

  // 1) conversions to bf16
  convert_vis<<<4096, 256, 0, stream>>>(x, vis_bf);
  convert_f2bf<<<2048, 256, 0, stream>>>(text, text_bf, (size_t)B_ * Lt_ * Ct_);
  convert_f2bf<<<512,  256, 0, stream>>>(Wq,   Wq_bf,   (size_t)C_ * C_);
  convert_f2bf<<<512,  256, 0, stream>>>(Wk,   Wk_bf,   (size_t)C_ * Ct_);
  convert_f2bf<<<512,  256, 0, stream>>>(Wv,   Wv_bf,   (size_t)C_ * Ct_);
  convert_f2bf<<<512,  256, 0, stream>>>(W1,   W1_bf,   (size_t)C_ * C_);
  convert_f2bf<<<256,  256, 0, stream>>>(Wdpp, Wdpp_bf, (size_t)DPP_ * C_);

  // 2) projections (WMMA bf16, f32 accumulate; 16x64 tile per wave)
  gemm_bf16_wmma<<<cdiv((ROWS / 16) * (C_ / 64), 8), 256, 0, stream>>>(
      vis_bf, C_, Wq_bf, C_, bq, q_f, nullptr, C_, ROWS, C_, C_, 0);
  gemm_bf16_wmma<<<cdiv((B_ * Lt_ / 16) * (C_ / 64), 8), 256, 0, stream>>>(
      text_bf, Ct_, Wk_bf, Ct_, bk, k_f, nullptr, C_, B_ * Lt_, C_, Ct_, 0);
  gemm_bf16_wmma<<<cdiv((B_ * Lt_ / 16) * (C_ / 64), 8), 256, 0, stream>>>(
      text_bf, Ct_, Wv_bf, Ct_, bv, v_f, nullptr, C_, B_ * Lt_, C_, Ct_, 0);

  // 3) attention + residual -> fused (bf16)
  attn_kernel<<<dim3(H_, B_), 256, 0, stream>>>(q_f, k_f, v_f, x, fused_bf);

  // 4) hmid = GELU-sigmoid(fused @ W1^T + b1)  (WMMA, bf16 out)
  gemm_bf16_wmma<<<cdiv((ROWS / 16) * (C_ / 64), 8), 256, 0, stream>>>(
      fused_bf, C_, W1_bf, C_, b1, nullptr, hmid_bf, C_, ROWS, C_, C_, 1);

  // 5) logits + gumbel + policy + keep_prob
  logits_kernel<<<ROWS / 8, 256, 0, stream>>>(hmid_bf, W2, b2, gum, out,
                                              keep_prob, sums);

  // 6) per-batch DPP weights
  w_kernel<<<B_, 256, 0, stream>>>(keep_prob, wrow);

  // 7) phi = vis @ Wdpp^T  (WMMA; overlays dead q buffer)
  gemm_bf16_wmma<<<cdiv((ROWS / 16) * (DPP_ / 64), 8), 256, 0, stream>>>(
      vis_bf, C_, Wdpp_bf, C_, nullptr, phi, nullptr, DPP_, ROWS, DPP_, C_, 0);

  // 8) normalize rows, scale by w -> pw (bf16)
  pw_kernel<<<ROWS / 8, 256, 0, stream>>>(phi, wrow, pw);

  // 9) Gram matrices (WMMA, guarded 13x13 tiles; overlays dead k/v)
  gram_wmma<<<dim3(B_, cdiv(13 * 13, 8)), 256, 0, stream>>>(pw, Gm);

  // 10) Cholesky in place (L2-resident), accumulate logdet
  cholesky_kernel<<<B_, 256, 0, stream>>>(Gm, sums);

  // 11) scalar losses
  finalize_kernel<<<1, 1, 0, stream>>>(sums, out);
}
